// StackedLSTM_68461778698469
// MI455X (gfx1250) — compile-verified, matrix operand order checked by reading, not run on hardware
//
#include <hip/hip_runtime.h>
#include <cstdint>
#include <cstddef>

// Problem constants (match reference): B=128, T=512, E=1024, H=1024, gates=4H=4096
#define B_SZ 128
#define T_SZ 512
#define E_SZ 1024
#define H_SZ 1024
#define G4_SZ 4096

typedef __attribute__((ext_vector_type(16))) __bf16 v16bf;
typedef __attribute__((ext_vector_type(8)))  float  v8f;

union Frag16 {
  v16bf v;
  uint4 u[2];
};

__device__ __forceinline__ unsigned short f2bf(float x) {
  unsigned int u = __float_as_uint(x);
  u += 0x7FFFu + ((u >> 16) & 1u);   // round-to-nearest-even
  return (unsigned short)(u >> 16);
}

// ---------------------------------------------------------------------------
// Prep kernel: fp32 -> bf16 bulk convert (for X)
// ---------------------------------------------------------------------------
__global__ __launch_bounds__(256) void convert_x_bf16(const float* __restrict__ in,
                                                      unsigned short* __restrict__ out,
                                                      size_t n) {
  size_t base = (size_t)blockIdx.x * 1024 + threadIdx.x;
#pragma unroll
  for (int r = 0; r < 4; ++r) {
    size_t j = base + (size_t)r * 256;
    if (j < n) out[j] = f2bf(in[j]);
  }
}

// ---------------------------------------------------------------------------
// Prep kernel: pack a (1024 x 4096) fp32 weight into bf16 WMMA B-fragment
// order. Fragment (nt, kt) covers N = nt*16..+15, K = kt*32..+31, stored as
// 512 contiguous bf16: lane L holds 16 values for column n = nt*16 + (L&15),
// K order = [0..7,16..23] (lanes 0-15) / [8..15,24..31] (lanes 16-31),
// mirroring the 16-bit A-fragment layout transposed (CDNA5 ISA).
// ---------------------------------------------------------------------------
__global__ __launch_bounds__(256) void pack_weight_bf16(const float* __restrict__ W,
                                                        unsigned short* __restrict__ out) {
  unsigned tid  = blockIdx.x * 256u + threadIdx.x;   // < 4,194,304
  unsigned f    = tid >> 9;        // fragment index = nt*32 + kt
  unsigned p    = tid & 511u;
  unsigned lane = p >> 4;
  unsigned j    = p & 15u;
  unsigned nt   = f >> 5;
  unsigned kt   = f & 31u;
  unsigned n    = nt * 16u + (lane & 15u);
  unsigned k    = kt * 32u + ((j < 8u) ? j : j + 8u) + ((lane < 16u) ? 0u : 8u);
  out[tid] = f2bf(W[(size_t)k * G4_SZ + n]);
}

// ---------------------------------------------------------------------------
// Zero-init kernel for state buffers (and the global-barrier counter)
// ---------------------------------------------------------------------------
__global__ __launch_bounds__(256) void zero_f32(float* __restrict__ p, size_t n) {
  size_t i = (size_t)blockIdx.x * 256 + threadIdx.x;
  if (i < n) p[i] = 0.0f;
}

// ---------------------------------------------------------------------------
// K=1024 GEMM slab: acc[q] += A(16x1024 rows of this wave) @ Wq(1024 x 16)
// for the four gate column-tiles q. One A fragment feeds 4 WMMAs.
// No pointer arrays anywhere -> addresses provably global -> global_load_b128.
// ---------------------------------------------------------------------------
__device__ __forceinline__ void gemm_k1024(const unsigned short* __restrict__ A,
                                           size_t astride,
                                           const unsigned short* __restrict__ Wp,
                                           v8f acc[4],
                                           int wg, int lane, int rowA, int laneK) {
  for (int kt = 0; kt < 32; ++kt) {
    Frag16 a;
    const unsigned short* ap = A + (size_t)rowA * astride + (unsigned)(kt * 32 + laneK);
    a.u[0] = *(const uint4*)(ap);                 // K .. K+7
    a.u[1] = *(const uint4*)(ap + 16);            // K+16 .. K+23
#pragma unroll
    for (int q = 0; q < 4; ++q) {                 // four gate column-tiles
      const unsigned short* bp =
          Wp + (((size_t)(q * 64 + wg) * 32 + (size_t)kt) << 9) + lane * 16;
      Frag16 b;
      b.u[0] = *(const uint4*)(bp);
      b.u[1] = *(const uint4*)(bp + 8);
      // Prefetch next kt fragment at WGP scope (default SCOPE=0: pull into all
      // cache levels). __builtin_prefetch can only reach SE/SYS scope.
      asm volatile("global_prefetch_b8 %0, off offset:1024" :: "v"(bp));
      acc[q] = __builtin_amdgcn_wmma_f32_16x16x32_bf16(
          false, a.v, false, b.v, (short)0, acc[q], false, false);
    }
  }
}

// ---------------------------------------------------------------------------
// One LSTM cell step for one layer / one 16-column tile:
//   pre = A1 @ W1 + A2 @ W2 + bias   (A1 = x_t or h0_t; A2 = h_prev, K=1024 ea)
// ---------------------------------------------------------------------------
__device__ __forceinline__ void lstm_cell_step(
    const unsigned short* __restrict__ A1, size_t a1_stride,
    const unsigned short* __restrict__ Wp1,
    const unsigned short* __restrict__ A2,          // stride H_SZ
    const unsigned short* __restrict__ Wp2,
    const float* __restrict__ b_f, const float* __restrict__ b_iog,
    const float* __restrict__ mask, int t,
    float* __restrict__ c_st, float* __restrict__ h_f32,
    unsigned short* __restrict__ h_out_bf,
    float* __restrict__ seq_out,                    // nullptr for layer 0
    int wg, int lane, int w)
{
  const int rowA  = w * 16 + (lane & 15);
  const int laneK = (lane < 16) ? 0 : 8;

  v8f acc[4] = { {}, {}, {}, {} };                  // f, i, o, g accumulators

  gemm_k1024(A1, a1_stride,   Wp1, acc, wg, lane, rowA, laneK);
  gemm_k1024(A2, (size_t)H_SZ, Wp2, acc, wg, lane, rowA, laneK);

  // Epilogue: C layout => lane%16 = column, VGPR r = row (r / r+8 halves)
  const int colH = wg * 16 + (lane & 15);
  const float bf = b_f[colH];
  const float bi = b_iog[colH];
  const float bo = b_iog[1024 + colH];
  const float bg = b_iog[2048 + colH];

#pragma unroll
  for (int r = 0; r < 8; ++r) {
    const int    row = w * 16 + r + ((lane < 16) ? 0 : 8);
    const size_t idx = (size_t)row * H_SZ + colH;
    const float  m   = mask[(size_t)row * T_SZ + t];

    float fg = acc[0][r] + bf;
    float ig = acc[1][r] + bi;
    float og = acc[2][r] + bo;
    float gg = acc[3][r] + bg;

    float cp = c_st[idx];
    float hp = h_f32[idx];

    float sf = 1.0f / (1.0f + __expf(-fg));
    float si = 1.0f / (1.0f + __expf(-ig));
    float so = 1.0f / (1.0f + __expf(-og));

    float c1 = sf * cp + si * tanhf(gg);
    c1 = c1 * m + cp * (1.0f - m);
    float h1 = so * tanhf(c1);
    h1 = h1 * m + hp * (1.0f - m);

    c_st[idx]     = c1;
    h_f32[idx]    = h1;
    h_out_bf[idx] = f2bf(h1);
    if (seq_out)  seq_out[((size_t)row * T_SZ + t) * H_SZ + colH] = h1;
  }
}

// ---------------------------------------------------------------------------
// Persistent kernel: 128 WGs = {layer 0, layer 1} x 64 column tiles.
// Round r: layer 0 computes timestep r while layer 1 computes timestep r-1
// (software pipeline across layers). Rounds separated by a device-scope
// arrival barrier (atomic counter + acq/rel fences, s_sleep spin).
// 128 WGs x 8 waves, no LDS -> trivially co-resident on MI455X.
// Ping-pong selection uses ternaries over kernel args (keeps loads global).
// ---------------------------------------------------------------------------
__global__ __launch_bounds__(256) void lstm_persistent(
    const unsigned short* __restrict__ x_bf,
    const unsigned short* __restrict__ Wp_ih0, const unsigned short* __restrict__ Wp_hh0,
    const unsigned short* __restrict__ Wp_ih1, const unsigned short* __restrict__ Wp_hh1,
    unsigned short* __restrict__ h0p0, unsigned short* __restrict__ h0p1,
    unsigned short* __restrict__ h1p0, unsigned short* __restrict__ h1p1,
    float* __restrict__ c0, float* __restrict__ c1s,
    float* __restrict__ h0f, float* __restrict__ h1f,
    const float* __restrict__ b_f0, const float* __restrict__ b_iog0,
    const float* __restrict__ b_f1, const float* __restrict__ b_iog1,
    const float* __restrict__ mask,
    float* __restrict__ out,
    unsigned* __restrict__ barrier_ctr)
{
  const int lane  = threadIdx.x & 31;
  const int w     = threadIdx.x >> 5;
  const int layer = (int)(blockIdx.x >> 6);       // 0 or 1
  const int wg    = (int)(blockIdx.x & 63u);      // hidden column tile
  const unsigned nwg = gridDim.x;                 // 128

  for (int r = 0; r < T_SZ + 1; ++r) {
    if (layer == 0) {
      if (r < T_SZ) {
        const int t = r;
        const unsigned short* A2   = (t & 1) ? h0p1 : h0p0;   // h0_prev
        unsigned short*       hout = (t & 1) ? h0p0 : h0p1;   // h0_t
        lstm_cell_step(x_bf + (size_t)t * E_SZ, (size_t)T_SZ * E_SZ, Wp_ih0,
                       A2, Wp_hh0, b_f0, b_iog0, mask, t,
                       c0, h0f, hout, (float*)nullptr, wg, lane, w);
      }
    } else {
      if (r >= 1) {
        const int t = r - 1;
        const unsigned short* A1   = (t & 1) ? h0p0 : h0p1;   // h0_t (from last round)
        const unsigned short* A2   = (t & 1) ? h1p1 : h1p0;   // h1_prev
        unsigned short*       hout = (t & 1) ? h1p0 : h1p1;   // h1_t
        lstm_cell_step(A1, (size_t)H_SZ, Wp_ih1,
                       A2, Wp_hh1, b_f1, b_iog1, mask, t,
                       c1s, h1f, hout, out, wg, lane, w);
      }
    }

    // ---- device-scope round barrier ----
    __threadfence();                               // release our h/c stores
    __syncthreads();
    if (threadIdx.x == 0) {
      __hip_atomic_fetch_add(barrier_ctr, 1u, __ATOMIC_RELEASE,
                             __HIP_MEMORY_SCOPE_AGENT);
      const unsigned target = (unsigned)(r + 1) * nwg;
      while (__hip_atomic_load(barrier_ctr, __ATOMIC_ACQUIRE,
                               __HIP_MEMORY_SCOPE_AGENT) < target) {
        __builtin_amdgcn_s_sleep(2);
      }
    }
    __syncthreads();
    __threadfence();                               // acquire: see peers' h stores
  }
}

// ---------------------------------------------------------------------------
// Host side
// ---------------------------------------------------------------------------
extern "C" void kernel_launch(void* const* d_in, const int* in_sizes, int n_in,
                              void* d_out, int out_size, void* d_ws, size_t ws_size,
                              hipStream_t stream) {
  (void)in_sizes; (void)n_in; (void)out_size; (void)ws_size;

  const float* x      = (const float*)d_in[0];
  const float* mask   = (const float*)d_in[1];
  const float* W_ih0  = (const float*)d_in[2];
  const float* W_hh0  = (const float*)d_in[3];
  const float* b_f0   = (const float*)d_in[4];
  const float* b_iog0 = (const float*)d_in[5];
  const float* W_ih1  = (const float*)d_in[6];
  const float* W_hh1  = (const float*)d_in[7];
  const float* b_f1   = (const float*)d_in[8];
  const float* b_iog1 = (const float*)d_in[9];
  float* out = (float*)d_out;

  char* p = (char*)d_ws;
  auto take = [&](size_t bytes) -> char* {
    char* r = p;
    p += (bytes + 255) & ~(size_t)255;
    return r;
  };

  const size_t wbytes = (size_t)E_SZ * G4_SZ * 2;            // 8 MB each (bf16)
  unsigned short* x_bf   = (unsigned short*)take((size_t)B_SZ * T_SZ * E_SZ * 2);
  unsigned short* Wp_ih0 = (unsigned short*)take(wbytes);
  unsigned short* Wp_hh0 = (unsigned short*)take(wbytes);
  unsigned short* Wp_ih1 = (unsigned short*)take(wbytes);
  unsigned short* Wp_hh1 = (unsigned short*)take(wbytes);

  // State buffers + barrier counter (contiguous so one zero pass covers all)
  const size_t hbf_bytes = (size_t)B_SZ * H_SZ * 2;          // 256 KB
  const size_t f32_bytes = (size_t)B_SZ * H_SZ * 4;          // 512 KB
  char* state_base = p;
  unsigned short* h0p0 = (unsigned short*)take(hbf_bytes);
  unsigned short* h0p1 = (unsigned short*)take(hbf_bytes);
  unsigned short* h1p0 = (unsigned short*)take(hbf_bytes);
  unsigned short* h1p1 = (unsigned short*)take(hbf_bytes);
  float* c0  = (float*)take(f32_bytes);
  float* c1s = (float*)take(f32_bytes);
  float* h0f = (float*)take(f32_bytes);
  float* h1f = (float*)take(f32_bytes);
  unsigned* bar = (unsigned*)take(256);
  const size_t state_floats = (size_t)(p - state_base) / 4;

  // --- prep: convert X, pack weights, zero state (incl. barrier counter) ---
  {
    const size_t n = (size_t)B_SZ * T_SZ * E_SZ;
    convert_x_bf16<<<(unsigned)((n + 1023) / 1024), 256, 0, stream>>>(x, x_bf, n);
  }
  {
    const unsigned blocks = (unsigned)(((size_t)E_SZ * G4_SZ) / 256);  // 16384
    pack_weight_bf16<<<blocks, 256, 0, stream>>>(W_ih0, Wp_ih0);
    pack_weight_bf16<<<blocks, 256, 0, stream>>>(W_hh0, Wp_hh0);
    pack_weight_bf16<<<blocks, 256, 0, stream>>>(W_ih1, Wp_ih1);
    pack_weight_bf16<<<blocks, 256, 0, stream>>>(W_hh1, Wp_hh1);
  }
  zero_f32<<<(unsigned)((state_floats + 255) / 256), 256, 0, stream>>>(
      (float*)state_base, state_floats);

  // --- one persistent kernel runs the whole recurrence (513 rounds) ---
  lstm_persistent<<<128, 256, 0, stream>>>(
      x_bf, Wp_ih0, Wp_hh0, Wp_ih1, Wp_hh1,
      h0p0, h0p1, h1p0, h1p1,
      c0, c1s, h0f, h1f,
      b_f0, b_iog0, b_f1, b_iog1,
      mask, out, bar);
}
